// TABlock_55284819034537
// MI455X (gfx1250) — matrix operand order repaired and checked by reading, not compiled, not run on hardware
//
#include <hip/hip_runtime.h>
#include <hip/hip_bf16.h>
#include <math.h>
#include <stdint.h>

#define BDIM 16
#define CTOK 1024
#define NDIM 768
#define MTOT (BDIM * CTOK) /* 16384 rows of X */

typedef __bf16 bf16_t;
typedef __bf16 v16bf __attribute__((ext_vector_type(16)));
typedef __bf16 v8bf  __attribute__((ext_vector_type(8)));
typedef float  v8f   __attribute__((ext_vector_type(8)));

__device__ __forceinline__ bf16_t f2bf(float f) {
  union { float f; uint32_t u; } in; in.f = f;
  uint32_t u = (in.u + 0x7FFFu + ((in.u >> 16) & 1u)) >> 16;  // RNE
  union { uint16_t s; bf16_t b; } out; out.s = (uint16_t)u;
  return out.b;
}

__device__ __forceinline__ v8f wmma_bf16(v16bf a, v16bf b, v8f c) {
  // D = A(16x32 bf16) * B(32x16 bf16) + C(16x16 f32)
  return __builtin_amdgcn_wmma_f32_16x16x32_bf16(false, a, false, b,
                                                 (short)0, c, false, false);
}

union AFrag { v16bf v; v8bf h[2]; };

// A fragment (16x32 bf16): lane L holds row L%16; halves at
// k0+(L/16)*16 and k0+(L/16)*16+8 (contiguous, matching B).
__device__ __forceinline__ v16bf load_a(const bf16_t* arow, int k0, int hl) {
  AFrag a;
  a.h[0] = *(const v8bf*)(arow + k0 + hl * 16);
  a.h[1] = *(const v8bf*)(arow + k0 + hl * 16 + 8);
  return a.v;
}

// 16(M) x 64(N) x K GEMM strip on one wave, 4 accum tiles, software-
// pipelined: next k-step's A + 4 B fragments are issued before the current
// k-step's WMMAs so the s_wait before each WMMA group only drains the older
// load group. Single base pointer per stream + constant immediate offsets
// (full unroll) -> no pointer arrays, no scratch spill.
template <int K, int LDB>
__device__ __forceinline__ void gemm_strip(const bf16_t* __restrict__ arow,
                                           const bf16_t* __restrict__ brow0,
                                           int hl, v8f acc[4]) {
  v16bf a  = load_a(arow, 0, hl);
  v16bf b0 = *(const v16bf*)(brow0 + 0 * 16 * LDB);
  v16bf b1 = *(const v16bf*)(brow0 + 1 * 16 * LDB);
  v16bf b2 = *(const v16bf*)(brow0 + 2 * 16 * LDB);
  v16bf b3 = *(const v16bf*)(brow0 + 3 * 16 * LDB);
#pragma unroll
  for (int k0 = 32; k0 < K; k0 += 32) {
    v16bf an  = load_a(arow, k0, hl);
    v16bf bn0 = *(const v16bf*)(brow0 + 0 * 16 * LDB + k0);
    v16bf bn1 = *(const v16bf*)(brow0 + 1 * 16 * LDB + k0);
    v16bf bn2 = *(const v16bf*)(brow0 + 2 * 16 * LDB + k0);
    v16bf bn3 = *(const v16bf*)(brow0 + 3 * 16 * LDB + k0);
    acc[0] = wmma_bf16(a, b0, acc[0]);
    acc[1] = wmma_bf16(a, b1, acc[1]);
    acc[2] = wmma_bf16(a, b2, acc[2]);
    acc[3] = wmma_bf16(a, b3, acc[3]);
    a = an; b0 = bn0; b1 = bn1; b2 = bn2; b3 = bn3;
  }
  acc[0] = wmma_bf16(a, b0, acc[0]);
  acc[1] = wmma_bf16(a, b1, acc[1]);
  acc[2] = wmma_bf16(a, b2, acc[2]);
  acc[3] = wmma_bf16(a, b3, acc[3]);
}

// ---------------------------------------------------------------- stage 1
// f32 -> bf16, 4 elements per thread (n must be a multiple of 4)
__global__ void cvt_f32_bf16_x4(const float* __restrict__ src,
                                bf16_t* __restrict__ dst, int n4) {
  int i = blockIdx.x * blockDim.x + threadIdx.x;
  if (i >= n4) return;
  float4 f = *(const float4*)(src + (size_t)i * 4);
  union { bf16_t b[4]; uint64_t u; } o;
  o.b[0] = f2bf(f.x); o.b[1] = f2bf(f.y);
  o.b[2] = f2bf(f.z); o.b[3] = f2bf(f.w);
  *(uint64_t*)(dst + (size_t)i * 4) = o.u;
}

// ---------------------------------------------------------------- stage 2
// Q/K: out[m*NDIM + d] = sum_k X[m,k]*W[d,k] + bias[d]   (row-major bf16)
// V  : out[(b*NDIM + d)*CTOK + c] = ...                  (transposed bf16)
__global__ void qkv_gemm(const bf16_t* __restrict__ X,
                         const bf16_t* __restrict__ W,
                         const float* __restrict__ bias,
                         bf16_t* __restrict__ out, int mode) {
  const int NG = NDIM / 64;                      // 12 n-groups of 64
  int wave = blockIdx.x * (blockDim.x >> 5) + (threadIdx.x >> 5);
  int lane = threadIdx.x & 31;
  int hl = lane >> 4, r = lane & 15;
  int mtile = wave / NG;
  int ng = wave % NG;
  if (mtile >= MTOT / 16) return;                // wave-uniform
  int n0 = ng * 64;

  const bf16_t* arow  = X + (size_t)(mtile * 16 + r) * NDIM;
  const bf16_t* brow0 = W + (size_t)(n0 + r) * NDIM + hl * 16;

  v8f acc[4] = {};
  gemm_strip<NDIM, NDIM>(arow, brow0, hl, acc);

#pragma unroll
  for (int t = 0; t < 4; ++t) {
    int d = n0 + t * 16 + r;
    float bv = bias[d];
#pragma unroll
    for (int rr = 0; rr < 8; ++rr) {
      int m = mtile * 16 + rr + hl * 8;
      float val = acc[t][rr] + bv;
      if (mode == 2) {                            // V: store transposed
        int b = m >> 10, c = m & (CTOK - 1);
        out[((size_t)b * NDIM + d) * CTOK + c] = f2bf(val);
      } else {
        out[(size_t)m * NDIM + d] = f2bf(val);
      }
    }
  }
}

// ---------------------------------------------------------------- stage 3
// S[b,c,e] = scale * sum_d Q[b,c,d]*K[b,e,d]   (f32)
__global__ void attn_scores(const bf16_t* __restrict__ Q,
                            const bf16_t* __restrict__ K,
                            float* __restrict__ S, float scale) {
  const int EG = CTOK / 64;                      // 16 e-groups
  int wave = blockIdx.x * (blockDim.x >> 5) + (threadIdx.x >> 5);
  int lane = threadIdx.x & 31;
  int hl = lane >> 4, r = lane & 15;
  int b = wave / ((CTOK / 16) * EG);
  int rem = wave % ((CTOK / 16) * EG);
  int ctile = rem / EG;
  int eg = rem % EG;
  if (b >= BDIM) return;
  int e0 = eg * 64;

  const bf16_t* arow  = Q + ((size_t)b * CTOK + ctile * 16 + r) * NDIM;
  const bf16_t* brow0 = K + ((size_t)b * CTOK + e0 + r) * NDIM + hl * 16;

  v8f acc[4] = {};
  gemm_strip<NDIM, NDIM>(arow, brow0, hl, acc);

#pragma unroll
  for (int t = 0; t < 4; ++t) {
    int e = e0 + t * 16 + r;
#pragma unroll
    for (int rr = 0; rr < 8; ++rr) {
      int c = ctile * 16 + rr + hl * 8;
      S[((size_t)b * CTOK + c) * CTOK + e] = acc[t][rr] * scale;
    }
  }
}

// ---------------------------------------------------------------- stage 4
// one wave per row of length CTOK; f32 in -> bf16 probabilities out
__global__ void softmax_rows(const float* __restrict__ S,
                             bf16_t* __restrict__ P) {
  int wave = blockIdx.x * (blockDim.x >> 5) + (threadIdx.x >> 5);
  int lane = threadIdx.x & 31;
  if (wave >= BDIM * CTOK) return;
  const float* row = S + (size_t)wave * CTOK;
  float4 v[CTOK / 128];
  float mx = -3.4e38f;
#pragma unroll
  for (int i = 0; i < CTOK / 128; ++i) {
    v[i] = *(const float4*)(row + i * 128 + lane * 4);
    mx = fmaxf(mx, fmaxf(fmaxf(v[i].x, v[i].y), fmaxf(v[i].z, v[i].w)));
  }
#pragma unroll
  for (int o = 16; o > 0; o >>= 1) mx = fmaxf(mx, __shfl_xor(mx, o, 32));
  float sum = 0.f;
#pragma unroll
  for (int i = 0; i < CTOK / 128; ++i) {
    v[i].x = __expf(v[i].x - mx); v[i].y = __expf(v[i].y - mx);
    v[i].z = __expf(v[i].z - mx); v[i].w = __expf(v[i].w - mx);
    sum += v[i].x + v[i].y + v[i].z + v[i].w;
  }
#pragma unroll
  for (int o = 16; o > 0; o >>= 1) sum += __shfl_xor(sum, o, 32);
  float inv = 1.0f / sum;
  bf16_t* prow = P + (size_t)wave * CTOK;
#pragma unroll
  for (int i = 0; i < CTOK / 128; ++i) {
    union { bf16_t b[4]; uint64_t u; } o;
    o.b[0] = f2bf(v[i].x * inv); o.b[1] = f2bf(v[i].y * inv);
    o.b[2] = f2bf(v[i].z * inv); o.b[3] = f2bf(v[i].w * inv);
    *(uint64_t*)(prow + i * 128 + lane * 4) = o.u;
  }
}

// ---------------------------------------------------------------- stage 5
// O[b,c,n] = sum_e P[b,c,e] * VT[b,n,e]; write transposed-flat + residual:
// out[b*C*N + n*C + c] = O + x.flat[same]
__global__ void attn_out(const bf16_t* __restrict__ P,
                         const bf16_t* __restrict__ VT,
                         const float* __restrict__ x,
                         float* __restrict__ out) {
  const int NG = NDIM / 64;
  int wave = blockIdx.x * (blockDim.x >> 5) + (threadIdx.x >> 5);
  int lane = threadIdx.x & 31;
  int hl = lane >> 4, r = lane & 15;
  int b = wave / ((CTOK / 16) * NG);
  int rem = wave % ((CTOK / 16) * NG);
  int ctile = rem / NG;
  int ng = rem % NG;
  if (b >= BDIM) return;
  int n0 = ng * 64;

  const bf16_t* arow  = P + ((size_t)b * CTOK + ctile * 16 + r) * CTOK;
  const bf16_t* brow0 = VT + ((size_t)b * NDIM + n0 + r) * CTOK + hl * 16;

  v8f acc[4] = {};
  gemm_strip<CTOK, CTOK>(arow, brow0, hl, acc);

  size_t base = (size_t)b * CTOK * NDIM;
#pragma unroll
  for (int t = 0; t < 4; ++t) {
    int n = n0 + t * 16 + r;
#pragma unroll
    for (int rr = 0; rr < 8; ++rr) {
      int c = ctile * 16 + rr + hl * 8;
      size_t idx = base + (size_t)n * CTOK + c;
      out[idx] = acc[t][rr] + x[idx];
    }
  }
}

// ----------------------------------------------------------------
extern "C" void kernel_launch(void* const* d_in, const int* in_sizes, int n_in,
                              void* d_out, int out_size, void* d_ws,
                              size_t ws_size, hipStream_t stream) {
  const float* x  = (const float*)d_in[0];
  const float* Wq = (const float*)d_in[1];
  const float* bq = (const float*)d_in[2];
  const float* Wk = (const float*)d_in[3];
  const float* bk = (const float*)d_in[4];
  const float* Wv = (const float*)d_in[5];
  const float* bv = (const float*)d_in[6];
  float* out = (float*)d_out;

  char* p = (char*)d_ws;
  auto take = [&](size_t bytes) -> char* {
    char* q = p;
    p += (bytes + 255) & ~(size_t)255;
    return q;
  };
  bf16_t* xb  = (bf16_t*)take((size_t)MTOT * NDIM * 2);
  bf16_t* wqb = (bf16_t*)take((size_t)NDIM * NDIM * 2);
  bf16_t* wkb = (bf16_t*)take((size_t)NDIM * NDIM * 2);
  bf16_t* wvb = (bf16_t*)take((size_t)NDIM * NDIM * 2);
  bf16_t* qb  = (bf16_t*)take((size_t)MTOT * NDIM * 2);
  bf16_t* kb  = (bf16_t*)take((size_t)MTOT * NDIM * 2);
  bf16_t* vtb = (bf16_t*)take((size_t)MTOT * NDIM * 2);
  float*  S   = (float*)take((size_t)BDIM * CTOK * CTOK * 4);
  bf16_t* P   = (bf16_t*)take((size_t)BDIM * CTOK * CTOK * 2);

  const int nx4 = (MTOT * NDIM) / 4;
  const int nw4 = (NDIM * NDIM) / 4;
  cvt_f32_bf16_x4<<<(nx4 + 255) / 256, 256, 0, stream>>>(x, xb, nx4);
  cvt_f32_bf16_x4<<<(nw4 + 255) / 256, 256, 0, stream>>>(Wq, wqb, nw4);
  cvt_f32_bf16_x4<<<(nw4 + 255) / 256, 256, 0, stream>>>(Wk, wkb, nw4);
  cvt_f32_bf16_x4<<<(nw4 + 255) / 256, 256, 0, stream>>>(Wv, wvb, nw4);

  const int g1_waves = (MTOT / 16) * (NDIM / 64);          // 12288
  qkv_gemm<<<g1_waves / 8, 256, 0, stream>>>(xb, wqb, bq, qb, 0);
  qkv_gemm<<<g1_waves / 8, 256, 0, stream>>>(xb, wkb, bk, kb, 1);
  qkv_gemm<<<g1_waves / 8, 256, 0, stream>>>(xb, wvb, bv, vtb, 2);

  const float scale = 1.0f / sqrtf((float)NDIM);
  const int sc_waves = BDIM * (CTOK / 16) * (CTOK / 64);   // 16384
  attn_scores<<<sc_waves / 8, 256, 0, stream>>>(qb, kb, S, scale);

  softmax_rows<<<(BDIM * CTOK) / 8, 256, 0, stream>>>(S, P);

  const int o_waves = BDIM * (CTOK / 16) * (NDIM / 64);    // 12288
  attn_out<<<o_waves / 8, 256, 0, stream>>>(P, vtb, x, out);
}
